// SparseConvBlock_85220741087711
// MI455X (gfx1250) — matrix operand order repaired and checked by reading, not compiled, not run on hardware
//
#include <hip/hip_runtime.h>

typedef __attribute__((ext_vector_type(16))) _Float16 v16h;
typedef __attribute__((ext_vector_type(8)))  float    v8f;

union Frag16 { v16h h; uint4 q[2]; };

#define CIN  128
#define COUT 128
#define LDS_STRIDE 136   // 128 + 8 f16 pad -> 4-dword bank rotation per row

// ---------------- feats f32 -> f16 ----------------
__global__ void k_cvt_feats(const float* __restrict__ f, _Float16* __restrict__ o, int n2) {
    int i = blockIdx.x * blockDim.x + threadIdx.x;
    if (i >= n2) return;
    float2 v = ((const float2*)f)[i];
    union { _Float16 h[2]; unsigned u; } p;
    p.h[0] = (_Float16)v.x; p.h[1] = (_Float16)v.y;
    ((unsigned*)o)[i] = p.u;
}

// --------- W[k][c][d] f32 -> Wt[k][d][c] f16 (transpose for B frags) ---------
__global__ void k_cvt_w(const float* __restrict__ W, _Float16* __restrict__ wt, int total) {
    int e = blockIdx.x * blockDim.x + threadIdx.x;
    if (e >= total) return;
    int k = e >> 14;          // / (128*128)
    int r = e & 16383;
    int d = r >> 7;
    int c = r & 127;
    wt[e] = (_Float16)W[(k << 14) + (c << 7) + d];
}

// ---------------- main gather-WMMA-scatter ----------------
__global__ void __launch_bounds__(256)
k_spconv(const _Float16* __restrict__ feats16, const _Float16* __restrict__ wt16,
         const int* __restrict__ in_idx, const int* __restrict__ out_idx,
         const int* __restrict__ valid, float* __restrict__ out, int N) {
    __shared__ __align__(16) _Float16 ldsW[COUT * LDS_STRIDE];

    const int k = blockIdx.y;
    // Stage Wt[k] (128 x 128 f16) into LDS, padded rows. 2048 x 16B chunks.
    const _Float16* wk = wt16 + (size_t)k * (CIN * COUT);
    for (int i = threadIdx.x; i < 2048; i += blockDim.x) {
        int row = i >> 4;          // Cout row of Wt
        int ch  = i & 15;          // 8-f16 chunk within row
        *(uint4*)&ldsW[row * LDS_STRIDE + ch * 8] =
            *(const uint4*)&wk[row * CIN + ch * 8];
    }
    __syncthreads();

    const int wave  = threadIdx.x >> 5;
    const int lane  = threadIdx.x & 31;
    const int mBase = (blockIdx.x * (blockDim.x >> 5) + wave) * 16;
    if (mBase >= N) return;                 // uniform per-wave; no barriers below

    const int laneM = lane & 15;
    const bool hi   = lane >= 16;
    const int kOff  = k * N;

    // Per-lane gather row + mask (lane handles A-matrix row M = laneM)
    int m0 = mBase + laneM;
    int vld = 0, rowIdx = 0;
    if (m0 < N) { rowIdx = in_idx[kOff + m0]; vld = valid[kOff + m0]; }
    const _Float16* fr = feats16 + (size_t)rowIdx * CIN;

    v8f acc[8] = {};
    const uint4 z4 = make_uint4(0u, 0u, 0u, 0u);

#pragma unroll
    for (int kk = 0; kk < 4; ++kk) {
        // A frag: lane<16 holds K 0..15, lane>=16 holds K 16..31 (matches B)
        Frag16 a;
        int base = kk * 32 + (hi ? 16 : 0);
        if (vld) {
            a.q[0] = *(const uint4*)(fr + base);
            a.q[1] = *(const uint4*)(fr + base + 8);
        } else {
            a.q[0] = z4; a.q[1] = z4;
        }
#pragma unroll
        for (int t = 0; t < 8; ++t) {
            // B frag: lane = column N, lane<16 holds K 0..15, lane>=16 holds K 16..31
            Frag16 b;
            const uint4* p = (const uint4*)&ldsW[(t * 16 + laneM) * LDS_STRIDE + kk * 32 + (hi ? 16 : 0)];
            b.q[0] = p[0];
            b.q[1] = p[1];
            acc[t] = __builtin_amdgcn_wmma_f32_16x16x32_f16(
                false, a.h, false, b.h, (short)0, acc[t], false, false);
        }
    }

    // Scatter: C/D layout -> VGPR j : M = j + (hi?8:0), N = lane&15
    const int hi8 = hi ? 8 : 0;
    int dstRow[8], vj[8];
#pragma unroll
    for (int j = 0; j < 8; ++j) {
        int mj = mBase + j + hi8;
        if (mj < N) { dstRow[j] = out_idx[kOff + mj]; vj[j] = valid[kOff + mj]; }
        else        { dstRow[j] = 0;                  vj[j] = 0; }
    }
#pragma unroll
    for (int t = 0; t < 8; ++t) {
        int col = t * 16 + laneM;
#pragma unroll
        for (int j = 0; j < 8; ++j) {
            if (vj[j])
                atomicAdd(out + (size_t)dstRow[j] * COUT + col, acc[t][j]);
        }
    }
}

// ---------------- per-channel sum / sumsq ----------------
__global__ void k_reduce(const float* __restrict__ x, float* __restrict__ sums,
                         float* __restrict__ sumsq, int N, int rowsPer) {
    int c  = threadIdx.x;                 // 128 threads: one channel each
    int r0 = blockIdx.x * rowsPer;
    int r1 = r0 + rowsPer; if (r1 > N) r1 = N;
    float s = 0.f, ss = 0.f;
    for (int r = r0; r < r1; ++r) {
        float v = x[(size_t)r * COUT + c];
        s += v; ss += v * v;
    }
    atomicAdd(&sums[c], s);
    atomicAdd(&sumsq[c], ss);
}

// ---------------- BN (batch stats) + exact GELU, in place ----------------
__global__ void k_bn_gelu(float* __restrict__ x, const float* __restrict__ sums,
                          const float* __restrict__ sumsq, const float* __restrict__ gamma,
                          const float* __restrict__ beta, int total, float invN) {
    int i = blockIdx.x * blockDim.x + threadIdx.x;
    if (i >= total) return;
    int c = i & (COUT - 1);
    float mean = sums[c] * invN;
    float var  = sumsq[c] * invN - mean * mean;
    float y = (x[i] - mean) * rsqrtf(var + 1e-5f) * gamma[c] + beta[c];
    x[i] = 0.5f * y * (1.0f + erff(y * 0.70710678118654752f));
}

extern "C" void kernel_launch(void* const* d_in, const int* in_sizes, int n_in,
                              void* d_out, int out_size, void* d_ws, size_t ws_size,
                              hipStream_t stream) {
    const float* feats = (const float*)d_in[0];
    const float* W     = (const float*)d_in[1];
    const float* gamma = (const float*)d_in[2];
    const float* beta  = (const float*)d_in[3];
    const int* in_idx  = (const int*)d_in[4];
    const int* out_idx = (const int*)d_in[5];
    const int* valid   = (const int*)d_in[6];
    float* out = (float*)d_out;

    const int N = in_sizes[0] / CIN;      // 40000
    const int K = in_sizes[4] / N;        // 27

    // Workspace layout
    char* ws = (char*)d_ws;
    size_t featsBytes = (size_t)N * CIN * sizeof(_Float16);
    size_t off1 = (featsBytes + 255) & ~(size_t)255;
    size_t wtBytes = (size_t)K * CIN * COUT * sizeof(_Float16);
    size_t off2 = (off1 + wtBytes + 255) & ~(size_t)255;
    _Float16* feats16 = (_Float16*)ws;
    _Float16* wt16    = (_Float16*)(ws + off1);
    float* sums       = (float*)(ws + off2);      // 128
    float* sumsq      = sums + COUT;              // 128

    hipMemsetAsync(d_out, 0, (size_t)N * COUT * sizeof(float), stream);
    hipMemsetAsync(sums, 0, 2 * COUT * sizeof(float), stream);

    // 1) feats -> f16
    {
        int n2 = N * CIN / 2;
        k_cvt_feats<<<(n2 + 255) / 256, 256, 0, stream>>>(feats, feats16, n2);
    }
    // 2) W -> transposed f16
    {
        int total = K * CIN * COUT;
        k_cvt_w<<<(total + 255) / 256, 256, 0, stream>>>(W, wt16, total);
    }
    // 3) gather-WMMA-scatter
    {
        int mTiles = (N + 15) / 16;                 // 2500
        dim3 grid((mTiles + 7) / 8, K);
        k_spconv<<<grid, 256, 0, stream>>>(feats16, wt16, in_idx, out_idx, valid, out, N);
    }
    // 4) channel stats
    {
        int blocks = 256;
        int rowsPer = (N + blocks - 1) / blocks;
        k_reduce<<<blocks, COUT, 0, stream>>>(out, sums, sumsq, N, rowsPer);
    }
    // 5) BN + GELU in place
    {
        int total = N * COUT;
        k_bn_gelu<<<(total + 255) / 256, 256, 0, stream>>>(out, sums, sumsq, gamma, beta,
                                                           total, 1.0f / (float)N);
    }
}
